// GatedAttention_30253749633483
// MI455X (gfx1250) — matrix operand order repaired and checked by reading, not compiled, not run on hardware
//
#include <hip/hip_runtime.h>

// ---------------------------------------------------------------------------
// Types for CDNA5 WMMA
// ---------------------------------------------------------------------------
typedef __attribute__((ext_vector_type(16))) __bf16 v16bf;
typedef __attribute__((ext_vector_type(8)))  float  v8f;

union FragCast { int4 i2[2]; v16bf v; };

__device__ inline unsigned short f2bf(float f) {
    unsigned int u = __float_as_uint(f);
    unsigned int r = (u + 0x7FFFu + ((u >> 16) & 1u)) >> 16;
    return (unsigned short)r;
}

__device__ inline v8f zero8() {
    v8f z;
#pragma unroll
    for (int r = 0; r < 8; ++r) z[r] = 0.f;
    return z;
}

// A-fragment (16x32 bf16, M x K). ISA 7.12.2: lanes 0-15 hold K=0..7 (vgpr0-3)
// and K=16..23 (vgpr4-7); lanes 16-31 hold K=8..15 and K=24..31.
// p = &A[row][k] + sel*8  ->  chunks at +0 and +16 elements.
__device__ inline v16bf frag_a(const unsigned short* p) {
    FragCast f;
    f.i2[0] = *(const int4*)(p);
    f.i2[1] = *(const int4*)(p + 16);
    return f.v;
}

// B-fragment (32x16 bf16, K x N) with B stored TRANSPOSED (N x K row-major).
// Lanes 0-15 hold K=0..7 and K=16..23 of column N=lane; lanes 16-31 hold K=8..15 and K=24..31.
// p = &Bt[col][k] + sel*8 -> chunks at +0 and +16 elements.
__device__ inline v16bf frag_b(const unsigned short* p) {
    FragCast f;
    f.i2[0] = *(const int4*)(p);
    f.i2[1] = *(const int4*)(p + 16);
    return f.v;
}

__device__ inline v8f wmma_bf16(v16bf a, v16bf b, v8f c) {
    return __builtin_amdgcn_wmma_f32_16x16x32_bf16(
        /*neg_a=*/false, a, /*neg_b=*/false, b,
        /*c_mod=*/(short)0, c, /*reuse_a=*/false, /*reuse_b=*/false);
}

// ---------------------------------------------------------------------------
// Generic WMMA GEMM:  C[b] = epilogue( A[b] (MxK, bf16) * Bt[b]^T (NxK, bf16) )
// Workgroup: 256 threads = 8 waves (4 M x 2 N), tile 256(M) x 128(N).
// Wave tile: 64 x 64 = 4x4 wmma accumulators (128 acc VGPRs), ~1:1 load:wmma.
// Requires M%256==0, N%128==0, K%32==0, ksplit%32==0 (true for all uses here).
// MODE 0: +bias, ReLU  -> bf16
// MODE 1: *scale       -> f32
// MODE 2: (none)       -> bf16
// MODE 3: +bias, sigmoid(x)*tanh(x) -> f32
// Concat-K: k < ksplit reads A0, else A1 at (k - ksplit); same lda for both.
// NOTE: addresses are built as (kernel-arg-rooted base) + k index so clang's
// address-space inference keeps these as GLOBAL loads (not FLAT).
// ---------------------------------------------------------------------------
template <int MODE>
__global__ __launch_bounds__(256) void gemm_wmma(
    const unsigned short* __restrict__ A0, const unsigned short* __restrict__ A1,
    int ksplit, const unsigned short* __restrict__ Bt,
    const float* __restrict__ bias, void* __restrict__ Cout,
    int M, int N, int K, int lda, int ldb, int ldc,
    long long sA, long long sB, long long sC, float scale) {
    const int lane  = threadIdx.x & 31;
    const int wave  = threadIdx.x >> 5;
    const int l15   = lane & 15;
    const int sel   = lane >> 4;
    const int waveM = wave >> 1;   // 0..3
    const int waveN = wave & 1;    // 0..1
    const int m0 = blockIdx.y * 256 + waveM * 64;
    const int n0 = blockIdx.x * 128 + waveN * 64;
    const int b  = blockIdx.z;

    const unsigned short* A0b = A0 + (size_t)b * sA;
    const unsigned short* A1b = A1 + (size_t)b * sA;
    const unsigned short* Btb = Bt + (size_t)b * sB;

    v8f acc[4][4];
#pragma unroll
    for (int i = 0; i < 4; ++i)
#pragma unroll
        for (int j = 0; j < 4; ++j) acc[i][j] = zero8();

    // Loop-invariant per-lane B fragment bases (k added in the loop).
    const unsigned short* b0 = Btb + (size_t)(n0 +  0 + l15) * ldb + (sel << 3);
    const unsigned short* b1 = Btb + (size_t)(n0 + 16 + l15) * ldb + (sel << 3);
    const unsigned short* b2 = Btb + (size_t)(n0 + 32 + l15) * ldb + (sel << 3);
    const unsigned short* b3 = Btb + (size_t)(n0 + 48 + l15) * ldb + (sel << 3);

    const int kmid = (ksplit < K) ? ksplit : K;
#pragma unroll 1
    for (int phase = 0; phase < 2; ++phase) {
        const unsigned short* Ab = phase ? A1b : A0b;
        const int kbeg = phase ? kmid : 0;
        const int kend = phase ? K : kmid;
        const int kadj = phase ? -ksplit : 0;  // A-local k = k + kadj
        if (kbeg >= kend) continue;

        const unsigned short* a0 = Ab + (size_t)(m0 +  0 + l15) * lda + (sel << 3) + kadj;
        const unsigned short* a1 = Ab + (size_t)(m0 + 16 + l15) * lda + (sel << 3) + kadj;
        const unsigned short* a2 = Ab + (size_t)(m0 + 32 + l15) * lda + (sel << 3) + kadj;
        const unsigned short* a3 = Ab + (size_t)(m0 + 48 + l15) * lda + (sel << 3) + kadj;

        for (int k = kbeg; k < kend; k += 32) {
            v16bf af0 = frag_a(a0 + k);
            v16bf af1 = frag_a(a1 + k);
            v16bf af2 = frag_a(a2 + k);
            v16bf af3 = frag_a(a3 + k);
            v16bf bf0 = frag_b(b0 + k);
            v16bf bf1 = frag_b(b1 + k);
            v16bf bf2 = frag_b(b2 + k);
            v16bf bf3 = frag_b(b3 + k);

            acc[0][0] = wmma_bf16(af0, bf0, acc[0][0]);
            acc[0][1] = wmma_bf16(af0, bf1, acc[0][1]);
            acc[0][2] = wmma_bf16(af0, bf2, acc[0][2]);
            acc[0][3] = wmma_bf16(af0, bf3, acc[0][3]);
            acc[1][0] = wmma_bf16(af1, bf0, acc[1][0]);
            acc[1][1] = wmma_bf16(af1, bf1, acc[1][1]);
            acc[1][2] = wmma_bf16(af1, bf2, acc[1][2]);
            acc[1][3] = wmma_bf16(af1, bf3, acc[1][3]);
            acc[2][0] = wmma_bf16(af2, bf0, acc[2][0]);
            acc[2][1] = wmma_bf16(af2, bf1, acc[2][1]);
            acc[2][2] = wmma_bf16(af2, bf2, acc[2][2]);
            acc[2][3] = wmma_bf16(af2, bf3, acc[2][3]);
            acc[3][0] = wmma_bf16(af3, bf0, acc[3][0]);
            acc[3][1] = wmma_bf16(af3, bf1, acc[3][1]);
            acc[3][2] = wmma_bf16(af3, bf2, acc[3][2]);
            acc[3][3] = wmma_bf16(af3, bf3, acc[3][3]);
        }
    }

    // Epilogue. C/D layout (ISA 7.12.2): vgpr r, lanes 0-15 -> row m0+r, col=lane;
    // lanes 16-31 -> row m0+8+r, col=lane-16.
#pragma unroll
    for (int j = 0; j < 4; ++j) {
        const int col = n0 + 16 * j + l15;
        float bv = 0.f;
        if (MODE == 0 || MODE == 3) bv = bias[col];
#pragma unroll
        for (int i = 0; i < 4; ++i) {
#pragma unroll
            for (int r = 0; r < 8; ++r) {
                const int row = m0 + 16 * i + r + sel * 8;
                float x = acc[i][j][r];
                if (MODE == 0) {
                    x += bv;
                    x = x > 0.f ? x : 0.f;
                    ((unsigned short*)Cout)[(size_t)b * sC + (size_t)row * ldc + col] = f2bf(x);
                } else if (MODE == 1) {
                    ((float*)Cout)[(size_t)b * sC + (size_t)row * ldc + col] = x * scale;
                } else if (MODE == 2) {
                    ((unsigned short*)Cout)[(size_t)b * sC + (size_t)row * ldc + col] = f2bf(x);
                } else {
                    x += bv;
                    const float sg = 1.f / (1.f + __expf(-x));
                    const float th = tanhf(x);
                    ((float*)Cout)[(size_t)b * sC + (size_t)row * ldc + col] = sg * th;
                }
            }
        }
    }
}

// ---------------------------------------------------------------------------
// f32 -> bf16 elementwise convert (n % 4 == 0)
// ---------------------------------------------------------------------------
__global__ void convert_bf16_kernel(const float* __restrict__ in,
                                    unsigned short* __restrict__ out, long long n) {
    long long i = ((long long)blockIdx.x * blockDim.x + threadIdx.x) * 4;
    if (i >= n) return;
    float4 f = *(const float4*)(in + i);
    unsigned long long pack = (unsigned long long)f2bf(f.x) |
                              ((unsigned long long)f2bf(f.y) << 16) |
                              ((unsigned long long)f2bf(f.z) << 32) |
                              ((unsigned long long)f2bf(f.w) << 48);
    *(unsigned long long*)(out + i) = pack;
}

// ---------------------------------------------------------------------------
// Tiled transpose + convert: in f32 [R x C] -> out bf16 [C x R]; R,C % 32 == 0.
// Block (32, 8); grid (C/32, R/32, batches).
// ---------------------------------------------------------------------------
__global__ __launch_bounds__(256) void transpose_convert_kernel(
    const float* __restrict__ in, unsigned short* __restrict__ out,
    int R, int C, long long inStride, long long outStride) {
    __shared__ float tile[32][33];
    const float* inB = in + (size_t)blockIdx.z * inStride;
    unsigned short* outB = out + (size_t)blockIdx.z * outStride;
    const int r0 = blockIdx.y * 32, c0 = blockIdx.x * 32;
    const int tx = threadIdx.x, ty = threadIdx.y;
#pragma unroll
    for (int i = 0; i < 32; i += 8)
        tile[ty + i][tx] = inB[(size_t)(r0 + ty + i) * C + c0 + tx];
    __syncthreads();
#pragma unroll
    for (int i = 0; i < 32; i += 8)
        outB[(size_t)(c0 + ty + i) * R + r0 + tx] = f2bf(tile[tx][ty + i]);
}

// ---------------------------------------------------------------------------
// Masked softmax over rows of 512; one wave per row. Writes bf16 P in place
// over the f32 score buffer (same row base; bf16 row = first 1024B of the
// 2048B f32 row, so P has a row stride of 1024 ushorts).
// ---------------------------------------------------------------------------
__global__ __launch_bounds__(256) void softmax_mask_kernel(
    float* __restrict__ att, const int* __restrict__ mask,
    int rowsPerBatch, int totalRows) {
    constexpr int LM = 512;
    const int wave = threadIdx.x >> 5;
    const int lane = threadIdx.x & 31;
    const int row = blockIdx.x * 8 + wave;
    if (row >= totalRows) return;
    const int batch = row / rowsPerBatch;
    const float* arow = att + (size_t)row * LM;
    unsigned short* prow = (unsigned short*)att + (size_t)row * (2 * LM);

    float v[LM / 32];
    float mx = -3.0e38f;
#pragma unroll
    for (int j = 0; j < LM / 32; ++j) {
        const int c = lane + 32 * j;
        float x = arow[c];
        x -= 1e30f * (1.f - (float)mask[batch * LM + c]);
        v[j] = x;
        mx = fmaxf(mx, x);
    }
#pragma unroll
    for (int o = 16; o >= 1; o >>= 1) mx = fmaxf(mx, __shfl_xor(mx, o, 32));
    float sum = 0.f;
#pragma unroll
    for (int j = 0; j < LM / 32; ++j) {
        v[j] = __expf(v[j] - mx);
        sum += v[j];
    }
#pragma unroll
    for (int o = 16; o >= 1; o >>= 1) sum += __shfl_xor(sum, o, 32);
    const float inv = 1.f / sum;
#pragma unroll
    for (int j = 0; j < LM / 32; ++j) prow[lane + 32 * j] = f2bf(v[j] * inv);
}

// ---------------------------------------------------------------------------
// Host-side launcher
// ---------------------------------------------------------------------------
extern "C" void kernel_launch(void* const* d_in, const int* in_sizes, int n_in,
                              void* d_out, int out_size, void* d_ws, size_t ws_size,
                              hipStream_t stream) {
    (void)in_sizes; (void)n_in; (void)out_size; (void)ws_size;
    constexpr int N = 8, Ld = 2048, Lm = 512, D = 1024, H = 1024;

    const float* input  = (const float*)d_in[0];  // [8,2048,1024]
    const float* memory = (const float*)d_in[1];  // [8,512,1024]
    const int*   mask   = (const int*)d_in[2];    // [8,512]
    const float* W_in1  = (const float*)d_in[3];  // [1024,1024]
    const float* b_in1  = (const float*)d_in[4];
    const float* W_mem1 = (const float*)d_in[5];  // [1024,1024]
    const float* b_mem1 = (const float*)d_in[6];
    const float* W_in2  = (const float*)d_in[7];  // [2048,1024]
    const float* b_in2  = (const float*)d_in[8];
    float* out = (float*)d_out;                   // [8,2048,1024] ++ [8,512,1024]

    char* ws = (char*)d_ws;
    unsigned short* input_bf  = (unsigned short*)(ws);                      // 33554432 B
    unsigned short* indot_bf  = (unsigned short*)(ws + 33554432ll);         // 33554432 B
    unsigned short* memdot_bf = (unsigned short*)(ws + 67108864ll);         //  8388608 B
    unsigned short* memory_bf = (unsigned short*)(ws + 75497472ll);         //  8388608 B
    unsigned short* memT_bf   = (unsigned short*)(ws + 83886080ll);         //  8388608 B
    unsigned short* Wt1       = (unsigned short*)(ws + 92274688ll);         //  2097152 B
    unsigned short* Wt2       = (unsigned short*)(ws + 94371840ll);         //  2097152 B
    unsigned short* Wt3       = (unsigned short*)(ws + 96468992ll);         //  4194304 B
    float*          att       = (float*)(ws + 100663296ll);                 // 33554432 B (aliases P bf16)
    unsigned short* outone_bf = (unsigned short*)(ws + 134217728ll);        // 33554432 B
    // total ws used: 167,772,160 bytes

    // 1) convert activations to bf16
    convert_bf16_kernel<<<(N * Ld * D) / 4 / 256, 256, 0, stream>>>(input, input_bf, (long long)N * Ld * D);
    convert_bf16_kernel<<<(N * Lm * D) / 4 / 256, 256, 0, stream>>>(memory, memory_bf, (long long)N * Lm * D);

    // 2) transpose+convert weights to bf16 N-major; memory -> memT (V^T)
    transpose_convert_kernel<<<dim3(H / 32, D / 32, 1), dim3(32, 8), 0, stream>>>(W_in1, Wt1, D, H, 0, 0);
    transpose_convert_kernel<<<dim3(H / 32, D / 32, 1), dim3(32, 8), 0, stream>>>(W_mem1, Wt2, D, H, 0, 0);
    transpose_convert_kernel<<<dim3(H / 32, (D + D) / 32, 1), dim3(32, 8), 0, stream>>>(W_in2, Wt3, 2 * D, H, 0, 0);
    transpose_convert_kernel<<<dim3(D / 32, Lm / 32, N), dim3(32, 8), 0, stream>>>(
        memory, memT_bf, Lm, D, (long long)Lm * D, (long long)D * Lm);

    // 3) input_dot = relu(input @ W_in1 + b_in1) -> bf16   [16384 x 1024]
    gemm_wmma<0><<<dim3(H / 128, (N * Ld) / 256, 1), 256, 0, stream>>>(
        input_bf, input_bf, D, Wt1, b_in1, indot_bf,
        N * Ld, H, D, D, D, H, 0, 0, 0, 1.f);

    // 4) memory_dot = relu(memory @ W_mem1 + b_mem1) -> bf16   [4096 x 1024]
    gemm_wmma<0><<<dim3(H / 128, (N * Lm) / 256, 1), 256, 0, stream>>>(
        memory_bf, memory_bf, D, Wt2, b_mem1, memdot_bf,
        N * Lm, H, D, D, D, H, 0, 0, 0, 1.f);

    // 5) att = (input_dot @ memory_dot^T) / 32 -> f32   batched [8] 2048 x 512
    gemm_wmma<1><<<dim3(Lm / 128, Ld / 256, N), 256, 0, stream>>>(
        indot_bf, indot_bf, H, memdot_bf, nullptr, att,
        Ld, Lm, H, H, H, Lm,
        (long long)Ld * H, (long long)Lm * H, (long long)Ld * Lm, 0.03125f);

    // 6) P = softmax(mask(att)) -> bf16 in place (row stride 1024 ushorts)
    softmax_mask_kernel<<<(N * Ld) / 8, 256, 0, stream>>>(att, mask, Ld, N * Ld);

    // 7) output_one = P @ memory -> bf16   batched [8] 2048 x 1024 (B^T = memT)
    gemm_wmma<2><<<dim3(D / 128, Ld / 256, N), 256, 0, stream>>>(
        (const unsigned short*)att, (const unsigned short*)att, Lm, memT_bf, nullptr, outone_bf,
        Ld, D, Lm, 2 * Lm /*lda: P row stride*/, Lm, D,
        (long long)Ld * 2 * Lm, (long long)D * Lm, (long long)Ld * D, 1.f);

    // 8) out = sigmoid(pre) * tanh(pre), pre = [input | output_one] @ W_in2 + b_in2
    gemm_wmma<3><<<dim3(H / 128, (N * Ld) / 256, 1), 256, 0, stream>>>(
        input_bf, outone_bf, D /*ksplit*/, Wt3, b_in2, out,
        N * Ld, H, 2 * D, D /*lda of each half*/, 2 * D, H, 0, 0, 0, 1.f);

    // 9) second tuple element: forward memory unchanged
    hipMemcpyAsync(out + (size_t)N * Ld * H, memory, (size_t)N * Lm * D * sizeof(float),
                   hipMemcpyDeviceToDevice, stream);
}
